// MMINet_6605659701720
// MI455X (gfx1250) — matrix operand order repaired and checked by reading, not compiled, not run on hardware
//
#include <hip/hip_runtime.h>
#include <hip/hip_bf16.h>

typedef __attribute__((ext_vector_type(16))) _Float16 v16h;
typedef __attribute__((ext_vector_type(8)))  float    v8f;

#define HIDDEN 128
#define SEQ    2048
#define BATCH  4096
#define HSTR   136   // halfs per LDS row (272B): padded so 16 rows hit distinct banks

union AFrag { v16h v; float4 f4[2]; };

__device__ __forceinline__ float fast_sigmoid(float z) {
    // v_exp_f32 + v_add + v_rcp_f32
    return __builtin_amdgcn_rcpf(1.0f + __expf(-z));
}

// ---- prep: h2h_w (f32, row-major 128x128) -> f16 row-major in workspace ----
__global__ void convert_w_kernel(const float* __restrict__ w,
                                 _Float16* __restrict__ wf16) {
    const int i = blockIdx.x * 256 + threadIdx.x;
    if (i < HIDDEN * HIDDEN) wf16[i] = (_Float16)w[i];
}

__global__ __launch_bounds__(256, 1)
void mminet_kernel(const float* __restrict__ x, const float* __restrict__ var,
                   const float* __restrict__ s0, const float* __restrict__ x2h_w,
                   const _Float16* __restrict__ wf16, const float* __restrict__ dnn1_w,
                   const float* __restrict__ dnn1_b, const float* __restrict__ dnn2_w,
                   const float* __restrict__ dnn2_b, const int* __restrict__ n_init_p,
                   float* __restrict__ out)
{
    __shared__ _Float16 hbuf[2][16 * HSTR];  // ping-pong h state, f16 row-major
    __shared__ float4   sdat[2][16];         // per-row {x0, dB, x2, -} for step t
    __shared__ float    redbuf[16];
    __shared__ float    swbuf;

    const int tid  = threadIdx.x;
    const int wave = tid >> 5;
    const int lane = tid & 31;
    const int lo   = lane & 15;
    const int hi   = lane >> 4;
    const int bg0  = blockIdx.x * 16;
    const int hid  = wave * 16 + lo;     // this lane's hidden-out / oper index
    const int nini = *n_init_p;

    // ---- per-lane constants ----
    const float x2h0 = x2h_w[hid*4+0], x2h1 = x2h_w[hid*4+1];
    const float x2h2 = x2h_w[hid*4+2], x2h3 = x2h_w[hid*4+3];
    const float d1w  = dnn1_w[hid];
    const float d2w  = dnn2_w[hid];
    const float rop  = 5.0f * (float)(hid + 1) * (1.0f / 128.0f);
    const float ropi = 1.0f / rop;

    float vconst[8], s1[8], lacc[8], pprev[8], x0prev[8];
    #pragma unroll
    for (int r = 0; r < 8; ++r) {
        const int bglob = bg0 + r + 8*hi;
        const float v0 = var[bglob*2+0], v1 = var[bglob*2+1];
        vconst[r] = v0 * x2h2 + v1 * x2h3;
        lacc[r] = 0.f; pprev[r] = 0.f; x0prev[r] = 0.f;
    }

    // ---- resident B-operand tiles, loaded directly as f16 (no f32 provenance) ----
    // B order used here (contiguous k, f16 32x16): lane%16 = N; lanes 0-15 K=0..15, lanes 16-31
    // K=16..31, 2 halfs per VGPR in K order -> 16 contiguous halfs of the row.
    v16h W[4];
    #pragma unroll
    for (int c = 0; c < 4; ++c)
        W[c] = *(const v16h*)(wf16 + hid * HIDDEN + 32*c + 16*hi);

    // ---- prologue: x(t=0) into sdat[0]; zero reduction buffers ----
    if (tid < 16) {
        redbuf[tid] = 0.f;
        if (tid == 0) swbuf = 0.f;
        const float* xp = x + (size_t)(bg0 + tid) * SEQ * 3;
        sdat[0][tid] = make_float4(xp[0], xp[1], xp[2], 0.f);
    }
    __syncthreads();

    // s1 init (op_step on s0), H_hyst0 partial reduction, sum(dnn2_w)
    #pragma unroll
    for (int r = 0; r < 8; ++r) {
        const int b = r + 8*hi;
        const float dB0 = sdat[0][b].y;
        float s = (dB0 + s0[(size_t)(bg0 + b) * HIDDEN + hid]) * ropi;
        s = fminf(fmaxf(s, -1.f), 1.f) * rop;
        s1[r] = s;
        atomicAdd(&redbuf[b], s * d1w);
    }
    if (hi == 0) atomicAdd(&swbuf, d2w);
    __syncthreads();

    const float d1w128 = dnn1_w[HIDDEN];
    const float d1w129 = dnn1_w[HIDDEN + 1];
    const float d1b    = dnn1_b[0];
    const float swinv  = 1.0f / swbuf;

    // h_init[b,k] = (x0 - H_hyst0[b]) / sum(dnn2_w)  -> hbuf[0]
    #pragma unroll
    for (int r = 0; r < 8; ++r) {
        const int b = r + 8*hi;
        const int bglob = bg0 + b;
        const float v0 = var[bglob*2+0], v1 = var[bglob*2+1];
        const float Hh0 = redbuf[b] + v0*d1w128 + v1*d1w129 + d1b;
        const float hinit = (sdat[0][b].x - Hh0) * swinv;
        hbuf[0][b*HSTR + hid] = (_Float16)hinit;
    }
    __syncthreads();

    // ---- main recurrence: one barrier per step, double-buffered h and x ----
    const int seqm1 = SEQ - 1;
    for (int t = 0; t < SEQ; ++t) {
        const int p = t & 1;

        // prefetch next step's x rows into the other sdat buffer
        if (tid < 16 && t < seqm1) {
            const float* xp = x + ((size_t)(bg0 + tid) * SEQ + (t + 1)) * 3;
            sdat[p ^ 1][tid] = make_float4(xp[0], xp[1], xp[2], 0.f);
        }

        float x0c[8], dBc[8];
        v8f cacc;
        #pragma unroll
        for (int r = 0; r < 8; ++r) {
            const float4 sd = sdat[p][r + 8*hi];
            x0c[r] = sd.x;
            dBc[r] = sd.y;
            cacc[r] = sd.x * x2h0 + sd.z * x2h1 + vconst[r];  // xin @ x2h_w^T
        }

        // Load ALL four A fragments, then force them simultaneously live with an
        // empty asm barrier so the allocator cannot coalesce them into one 8-VGPR
        // set (which would serialize ds_load -> wait dscnt 0 -> wmma per chunk).
        AFrag A0, A1, A2, A3;
        {
            const float4* ap = (const float4*)&hbuf[p][lo * HSTR + 16*hi];
            A0.f4[0] = ap[0];  A0.f4[1] = ap[1];    // K =  0.. 31 slice
            A1.f4[0] = ap[4];  A1.f4[1] = ap[5];    // K = 32.. 63 slice
            A2.f4[0] = ap[8];  A2.f4[1] = ap[9];    // K = 64.. 95 slice
            A3.f4[0] = ap[12]; A3.f4[1] = ap[13];   // K = 96..127 slice
        }
        asm volatile("" : "+v"(A0.v), "+v"(A1.v), "+v"(A2.v), "+v"(A3.v));

        // h_{t-1} @ h2h_w^T : 4 chained v_wmma_f32_16x16x32_f16 (K=128)
        cacc = __builtin_amdgcn_wmma_f32_16x16x32_f16(
            false, A0.v, false, W[0], (short)0, cacc, false, false);
        cacc = __builtin_amdgcn_wmma_f32_16x16x32_f16(
            false, A1.v, false, W[1], (short)0, cacc, false, false);
        cacc = __builtin_amdgcn_wmma_f32_16x16x32_f16(
            false, A2.v, false, W[2], (short)0, cacc, false, false);
        cacc = __builtin_amdgcn_wmma_f32_16x16x32_f16(
            false, A3.v, false, W[3], (short)0, cacc, false, false);

        // op_step is independent of the WMMA result: fills the WMMA->VALU
        // data-hazard window instead of v_nops
        if (t > 0) {
            #pragma unroll
            for (int r = 0; r < 8; ++r) {
                float s = (dBc[r] + s1[r]) * ropi;
                s1[r] = fminf(fmaxf(s, -1.f), 1.f) * rop;
            }
        }

        const float gate = (t > nini) ? 1.0f : 0.0f;
        #pragma unroll
        for (int r = 0; r < 8; ++r) {
            const float hv = fast_sigmoid(cacc[r]);
            const int b = r + 8*hi;
            hbuf[p ^ 1][b*HSTR + hid] = (_Float16)hv;
            const float pc = hv * d2w + s1[r] * d1w;    // out partial (this lane)
            lacc[r] += gate * (x0c[r] - x0prev[r]) * (pc + pprev[r]); // trapezoid
            pprev[r]  = pc;
            x0prev[r] = x0c[r];
        }
        __syncthreads();
    }

    // ---- epilogue: reduce lane partials per batch row, finalize Pv ----
    __syncthreads();
    if (tid < 16) redbuf[tid] = 0.f;
    __syncthreads();
    #pragma unroll
    for (int r = 0; r < 8; ++r)
        atomicAdd(&redbuf[r + 8*hi], lacc[r]);
    __syncthreads();

    if (tid < 16) {
        const int bglob = bg0 + tid;
        const float v0 = var[bglob*2+0], v1 = var[bglob*2+1];
        const float oc = v0*d1w128 + v1*d1w129 + d1b + dnn2_b[0];  // const part of out
        const float x0f = x[((size_t)bglob * SEQ + nini)  * 3];
        const float x0l = x[((size_t)bglob * SEQ + seqm1) * 3];
        const float N01 = 0.0642790612f;   // NORM[0][1]
        const float N10 = 0.115118525f;    // NORM[1][0]
        const float N11 = 12.2041107f;     // NORM[1][1]
        const float N20 = 5.16368866f;     // NORM[2][0]
        const float N21 = 0.268540382f;    // NORM[2][1]
        const float integ = 0.5f * N11 * N01 * redbuf[tid]
                          + (oc * N11 + N10) * N01 * (x0l - x0f);
        const float p10 = __expf((v0 * N21 + N20) * 2.302585093f); // 10^e
        out[bglob] = integ * p10;
    }
}

extern "C" void kernel_launch(void* const* d_in, const int* in_sizes, int n_in,
                              void* d_out, int out_size, void* d_ws, size_t ws_size,
                              hipStream_t stream) {
    (void)in_sizes; (void)n_in; (void)ws_size; (void)out_size;
    const float* x      = (const float*)d_in[0];
    const float* var    = (const float*)d_in[1];
    const float* s0     = (const float*)d_in[2];
    const float* x2h_w  = (const float*)d_in[3];
    const float* h2h_w  = (const float*)d_in[4];
    const float* dnn1_w = (const float*)d_in[5];
    const float* dnn1_b = (const float*)d_in[6];
    const float* dnn2_w = (const float*)d_in[7];
    const float* dnn2_b = (const float*)d_in[8];
    const int*   n_init = (const int*)d_in[9];
    float* out = (float*)d_out;
    _Float16* wf16 = (_Float16*)d_ws;   // 128*128*2 = 32 KB of workspace

    hipLaunchKernelGGL(convert_w_kernel, dim3((HIDDEN*HIDDEN + 255) / 256),
                       dim3(256), 0, stream, h2h_w, wf16);

    hipLaunchKernelGGL(mminet_kernel, dim3(BATCH / 16), dim3(256), 0, stream,
                       x, var, s0, x2h_w, wf16, dnn1_w, dnn1_b, dnn2_w, dnn2_b,
                       n_init, out);
}
